// DMPNN_30623116821204
// MI455X (gfx1250) — matrix operand order repaired and checked by reading, not compiled, not run on hardware
//
#include <hip/hip_runtime.h>
#include <hip/hip_bf16.h>

// ---------------- problem constants (from reference) ----------------
constexpr int HD   = 300;
constexpr int AF   = 133;
constexpr int BF   = 147;
constexpr int NA   = 100000;
constexpr int NB   = 200000;
constexpr int NBR  = 6;
constexpr int NM   = 5000;
constexpr int NFA  = 40000;
constexpr int NFB  = 80000;
constexpr int NFBR = 4;
constexpr int NFM  = 5000;
constexpr int DEPTH = 3;

constexpr int NTILE = 19;   // ceil(300/16)
constexpr int NPAD  = 304;  // 19*16

typedef __attribute__((ext_vector_type(16))) __bf16 v16bf;
typedef __attribute__((ext_vector_type(8)))  float  v8f;
typedef __attribute__((ext_vector_type(4), aligned(4))) float f4u;  // dword-aligned vec4

// ---------------------------------------------------------------------------
// One-shot weight prep: fp32 W[w_off:w_off+K, 0:300] (row-major, ld=300)
//   -> bf16 Wt[n][k], n-major, n padded to 304, k padded to Kpad, zero-filled.
// ---------------------------------------------------------------------------
__global__ void wprep_k(const float* __restrict__ W, int w_off, int K, int Kpad,
                        __bf16* __restrict__ Wt_g)
{
    long i = (long)blockIdx.x * blockDim.x + threadIdx.x;
    long total = (long)NPAD * Kpad;
    if (i >= total) return;
    int n = (int)(i / Kpad);
    int k = (int)(i - (long)n * Kpad);
    float v = (n < HD && k < K) ? W[(long)(w_off + k) * HD + n] : 0.0f;
    Wt_g[i] = (__bf16)v;                      // native RNE cvt
}

// ---------------------------------------------------------------------------
// Streaming WMMA GEMM: C[M,300] = act( A[M,K] @ W + add + bias )
// Pre-packed bf16 W^T async-copied global->LDS; each wave computes a
// 32-row x 304-col tile: every B fragment feeds TWO independent
// v_wmma_f32_16x16x32_bf16 ops (acc0/acc1), doubling math per DS load.
// M must be a multiple of 32 (true at all call sites).
// ---------------------------------------------------------------------------
__global__ __launch_bounds__(256)
void gemm_bf16_ws(const float* __restrict__ A, int lda, int M, int K, int Kpad,
                  const __bf16* __restrict__ Wt_g,
                  const float* __restrict__ add,
                  const float* __restrict__ bias,
                  float* __restrict__ C,
                  float* __restrict__ Cpre,
                  int relu)
{
    extern __shared__ __bf16 Wt[];            // [NPAD][Kpad]
    const int tid = threadIdx.x;

    // ---- async copy of pre-packed W tile into LDS (16B chunks) ----
    {
        const int nchunks = (NPAD * Kpad) >> 3;       // 8 bf16 = 16B per chunk
        unsigned lds_base = (unsigned)(uintptr_t)&Wt[0];
        for (int c = tid; c < nchunks; c += blockDim.x) {
            unsigned loff = lds_base + ((unsigned)c << 4);
            const __bf16* gp = Wt_g + ((long)c << 3);
            asm volatile("global_load_async_to_lds_b128 %0, %1, off"
                         :: "v"(loff), "v"(gp) : "memory");
        }
        asm volatile("s_wait_asynccnt 0x0" ::: "memory");
    }
    __syncthreads();

    const int wave = tid >> 5;
    const int lane = tid & 31;
    const int half = lane >> 4;               // 0 or 1
    const int l15  = lane & 15;
    const long r0  = ((long)blockIdx.x * 8 + wave) * 32;   // 32 rows per wave
    if (r0 >= M) return;                      // uniform per wave

    v8f acc0[NTILE], acc1[NTILE];
#pragma unroll
    for (int t = 0; t < NTILE; ++t) { acc0[t] = (v8f)0.0f; acc1[t] = (v8f)0.0f; }

    const float* Arow0 = A + (r0 + l15)      * (long)lda;
    const float* Arow1 = A + (r0 + 16 + l15) * (long)lda;

    auto mma_all = [&](const v16bf& a0, const v16bf& a1, int kk) {
#pragma unroll
        for (int t = 0; t < NTILE; ++t) {
            const __bf16* bp =
                &Wt[(t * 16 + l15) * Kpad + kk + half * 8];
            v16bf b;
#pragma unroll
            for (int s = 0; s < 8; ++s) { b[s] = bp[s]; b[8 + s] = bp[16 + s]; }
            acc0[t] = __builtin_amdgcn_wmma_f32_16x16x32_bf16(
                false, a0, false, b, (short)0, acc0[t], false, false);
            acc1[t] = __builtin_amdgcn_wmma_f32_16x16x32_bf16(
                false, a1, false, b, (short)0, acc1[t], false, false);
        }
    };

    auto load_frag_fast = [&](const float* Ar, int kk) -> v16bf {
        const float* p0 = Ar + kk + half * 8;
        const float* p1 = Ar + kk + 16 + half * 8;
        f4u x0 = *(const f4u*)p0;
        f4u x1 = *(const f4u*)(p0 + 4);
        f4u x2 = *(const f4u*)p1;
        f4u x3 = *(const f4u*)(p1 + 4);
        v16bf a;
#pragma unroll
        for (int j = 0; j < 4; ++j) {
            a[j]      = (__bf16)x0[j];
            a[4 + j]  = (__bf16)x1[j];
            a[8 + j]  = (__bf16)x2[j];
            a[12 + j] = (__bf16)x3[j];
        }
        return a;
    };
    auto load_frag_tail = [&](const float* Ar, int kk) -> v16bf {
        const int kb0 = kk + half * 8;
        const int kb1 = kk + 16 + half * 8;
        v16bf a;
#pragma unroll
        for (int j = 0; j < 8; ++j) {
            int k0 = kb0 + j, k1 = kb1 + j;
            a[j]     = (__bf16)((k0 < K) ? Ar[k0] : 0.0f);
            a[8 + j] = (__bf16)((k1 < K) ? Ar[k1] : 0.0f);
        }
        return a;
    };

    int kk = 0;
    for (; kk + 32 <= K; kk += 32) {          // full K steps, unguarded vec loads
        v16bf a0 = load_frag_fast(Arow0, kk);
        v16bf a1 = load_frag_fast(Arow1, kk);
        mma_all(a0, a1, kk);
    }
    for (; kk < Kpad; kk += 32) {             // at most one guarded tail step
        v16bf a0 = load_frag_tail(Arow0, kk);
        v16bf a1 = load_frag_tail(Arow1, kk);
        mma_all(a0, a1, kk);
    }

    // ---- epilogue: VGPR i of tile t -> row r0 (+16) + i + half*8, col t*16+l15 ----
#pragma unroll
    for (int t = 0; t < NTILE; ++t) {
        const int col = t * 16 + l15;
        if (col >= HD) continue;
#pragma unroll
        for (int i = 0; i < 8; ++i) {
            {
                const long row = r0 + i + half * 8;
                const long idx = row * HD + col;
                float v = acc0[t][i];
                if (add)  v += add[idx];
                if (bias) v += bias[col];
                if (Cpre) Cpre[idx] = v;
                C[idx] = relu ? fmaxf(v, 0.0f) : v;
            }
            {
                const long row = r0 + 16 + i + half * 8;
                const long idx = row * HD + col;
                float v = acc1[t][i];
                if (add)  v += add[idx];
                if (bias) v += bias[col];
                if (Cpre) Cpre[idx] = v;
                C[idx] = relu ? fmaxf(v, 0.0f) : v;
            }
        }
    }
}

// ---------------- glue kernels ----------------

// out[x,:] = sum_j msg[x2y[x,j], :]
__global__ void gather_sum_k(const float* __restrict__ msg,
                             const int* __restrict__ x2y, int deg,
                             float* __restrict__ out)
{
    const int x = blockIdx.x;
    __shared__ int idx[8];
    if ((int)threadIdx.x < deg) idx[threadIdx.x] = x2y[(long)x * deg + threadIdx.x];
    __syncthreads();
    for (int h = threadIdx.x; h < HD; h += blockDim.x) {
        float s = 0.0f;
        for (int j = 0; j < deg; ++j) s += msg[(long)idx[j] * HD + h];
        out[(long)x * HD + h] = s;
    }
}

// out[b,:] = amsg[b2a[b],:] - msg[b2revb[b],:]
__global__ void bond_diff_k(const float* __restrict__ amsg,
                            const float* __restrict__ msg,
                            const int* __restrict__ b2a,
                            const int* __restrict__ b2revb,
                            float* __restrict__ out)
{
    const int b = blockIdx.x;
    const int a = b2a[b], rb = b2revb[b];
    for (int h = threadIdx.x; h < HD; h += blockDim.x)
        out[(long)b * HD + h] = amsg[(long)a * HD + h] - msg[(long)rb * HD + h];
}

// out[b,:] = ffeat[a2frag[b2a[b]], :]
__global__ void frag_gather_k(const float* __restrict__ ffeat,
                              const int* __restrict__ a2frag,
                              const int* __restrict__ b2a,
                              float* __restrict__ out)
{
    const int b = blockIdx.x;
    const int fr = a2frag[b2a[b]];
    for (int h = threadIdx.x; h < HD; h += blockDim.x)
        out[(long)b * HD + h] = ffeat[(long)fr * HD + h];
}

// deterministic segment-mean over sorted x2mol (binary search per molecule)
__global__ void seg_mean_k(const float* __restrict__ ah,
                           const int* __restrict__ x2mol, int nx,
                           float* __restrict__ out)
{
    const int m = blockIdx.x;
    int lo = 0, hi = nx;
    while (lo < hi) { int mid = (lo + hi) >> 1; if (x2mol[mid] < m) lo = mid + 1; else hi = mid; }
    const int s = lo;
    hi = nx;
    while (lo < hi) { int mid = (lo + hi) >> 1; if (x2mol[mid] < m + 1) lo = mid + 1; else hi = mid; }
    const int e = lo;
    const float inv = (e > s) ? 1.0f / (float)(e - s) : 0.0f;
    for (int h = threadIdx.x; h < HD; h += blockDim.x) {
        float acc = 0.0f;
        for (int x = s; x < e; ++x) acc += ah[(long)x * HD + h];
        out[(long)m * HD + h] = acc * inv;
    }
}

__global__ void zero_k(float* __restrict__ p, long n)
{
    long i = (long)blockIdx.x * blockDim.x + threadIdx.x;
    if (i < n) p[i] = 0.0f;
}

// ---------------- driver ----------------
extern "C" void kernel_launch(void* const* d_in, const int* in_sizes, int n_in,
                              void* d_out, int out_size, void* d_ws, size_t ws_size,
                              hipStream_t stream)
{
    const float* f_atoms       = (const float*)d_in[0];
    const float* f_bonds       = (const float*)d_in[1];
    const float* f_frags_atoms = (const float*)d_in[2];
    const float* f_frags_bonds = (const float*)d_in[3];
    const float* W_i           = (const float*)d_in[4];
    const float* W_h           = (const float*)d_in[5];
    const float* W_fus         = (const float*)d_in[6];
    const float* b_fus         = (const float*)d_in[7];
    const float* W_o           = (const float*)d_in[8];
    const float* b_o           = (const float*)d_in[9];
    const int*   a2b           = (const int*)d_in[10];
    const int*   b2a           = (const int*)d_in[11];
    const int*   b2revb        = (const int*)d_in[12];
    const int*   atom2mol      = (const int*)d_in[13];
    const int*   frags_a2b     = (const int*)d_in[14];
    const int*   frags_b2a     = (const int*)d_in[15];
    const int*   frags_b2revb  = (const int*)d_in[16];
    const int*   frag_a2mol    = (const int*)d_in[17];
    const int*   a2frag        = (const int*)d_in[18];

    // ---- workspace carve-out: fp32 buffers ----
    float* ws     = (float*)d_ws;
    float* inp    = ws;                          // NB*HD
    float* msg    = inp    + (long)NB  * HD;     // NB*HD
    float* finp   = msg    + (long)NB  * HD;     // NFB*HD
    float* fmsg   = finp   + (long)NFB * HD;     // NFB*HD
    float* amsg   = fmsg   + (long)NFB * HD;     // NA*HD
    float* mbuf   = amsg   + (long)NA  * HD;     // NB*HD
    float* fexp   = mbuf   + (long)NB  * HD;     // NB*HD (reused for atom W_o-top)
    float* frah   = fexp   + (long)NB  * HD;     // NFA*HD
    float* fratop = frah   + (long)NFA * HD;     // NFA*HD
    float* wsend  = fratop + (long)NFA * HD;

    // ---- bf16 pre-packed weight tiles ----
    constexpr int KP160 = 160, KP320 = 320;
    __bf16* wi_t  = (__bf16*)wsend;                       // [304][160]
    __bf16* wh_t  = wi_t  + (long)NPAD * KP160;           // [304][320]
    __bf16* wfA_t = wh_t  + (long)NPAD * KP320;           // [304][320]
    __bf16* wfB_t = wfA_t + (long)NPAD * KP320;           // [304][320]
    __bf16* woT_t = wfB_t + (long)NPAD * KP320;           // [304][160]
    __bf16* woB_t = woT_t + (long)NPAD * KP160;           // [304][320]

    // d_out layout: mol_vecs [NM,HD] | atom_hiddens [NA,HD] | frags_feature [NFM+1,HD]
    float* out_mol   = (float*)d_out;
    float* out_ah    = out_mol + (long)NM * HD;
    float* out_ffeat = out_ah  + (long)NA * HD;

    auto wprep = [&](const float* W, int w_off, int K, __bf16* dst) {
        int Kpad = (K + 31) & ~31;
        long total = (long)NPAD * Kpad;
        wprep_k<<<(int)((total + 255) / 256), 256, 0, stream>>>(W, w_off, K, Kpad, dst);
    };
    auto gemm = [&](const float* A, int lda, int M, int K, const __bf16* Wt,
                    const float* add, const float* bias,
                    float* C, float* Cpre, int relu) {
        int Kpad = (K + 31) & ~31;
        size_t sh = (size_t)NPAD * Kpad * sizeof(__bf16);
        int blocks = (M + 255) / 256;                     // 8 waves x 32 rows
        gemm_bf16_ws<<<blocks, 256, sh, stream>>>(A, lda, M, K, Kpad, Wt,
                                                  add, bias, C, Cpre, relu);
    };

    // ---- pre-pack all weight slices (bf16, transposed, padded) ----
    wprep(W_i,   0,  BF, wi_t);
    wprep(W_h,   0,  HD, wh_t);
    wprep(W_fus, 0,  HD, wfA_t);
    wprep(W_fus, HD, HD, wfB_t);
    wprep(W_o,   0,  AF, woT_t);
    wprep(W_o,   AF, HD, woB_t);

    // ---- initial projections ----
    gemm(f_bonds, BF, NB, BF, wi_t, nullptr, nullptr, msg, inp, 1);          // inp, msg
    gemm(f_frags_bonds, BF, NFB, BF, wi_t, nullptr, nullptr, fmsg, finp, 1); // finp, fmsg
    gemm(f_frags_atoms, AF, NFA, AF, woT_t, nullptr, nullptr, fratop, nullptr, 0);

    // zero padding row 0 of frags_feature
    zero_k<<<(HD + 255) / 256, 256, 0, stream>>>(out_ffeat, (long)HD);

    for (int it = 0; it < DEPTH - 1; ++it) {
        // ---- fragment message aggregation: fmsg = relu(finp + m @ W_h) ----
        gather_sum_k<<<NFA, 256, 0, stream>>>(fmsg, frags_a2b, NFBR, amsg);
        bond_diff_k<<<NFB, 256, 0, stream>>>(amsg, fmsg, frags_b2a, frags_b2revb, mbuf);
        gemm(mbuf, HD, NFB, HD, wh_t, finp, nullptr, fmsg, nullptr, 1);

        // ---- fragment readout -> frags_feature rows 1..NFM ----
        gather_sum_k<<<NFA, 256, 0, stream>>>(fmsg, frags_a2b, NFBR, amsg);
        gemm(amsg, HD, NFA, HD, woB_t, fratop, b_o, frah, nullptr, 1);
        seg_mean_k<<<NFM, 256, 0, stream>>>(frah, frag_a2mol, NFA, out_ffeat + HD);

        // ---- fusion: msg = msg@Wf_top + fexp@Wf_bot + b_fus (no act) ----
        frag_gather_k<<<NB, 256, 0, stream>>>(out_ffeat, a2frag, b2a, fexp);
        gemm(msg,  HD, NB, HD, wfA_t, nullptr, nullptr, mbuf, nullptr, 0);
        gemm(fexp, HD, NB, HD, wfB_t, mbuf,    b_fus,   msg,  nullptr, 0);

        // ---- atom message aggregation: msg = relu(inp + m @ W_h) ----
        gather_sum_k<<<NA, 256, 0, stream>>>(msg, a2b, NBR, amsg);
        bond_diff_k<<<NB, 256, 0, stream>>>(amsg, msg, b2a, b2revb, mbuf);
        gemm(mbuf, HD, NB, HD, wh_t, inp, nullptr, msg, nullptr, 1);
    }
    // (post-loop fragment readout == last in-loop readout; out_ffeat already final)

    // ---- final atom readout ----
    gather_sum_k<<<NA, 256, 0, stream>>>(msg, a2b, NBR, amsg);
    gemm(f_atoms, AF, NA, AF, woT_t, nullptr, nullptr, fexp /*atom top*/, nullptr, 0);
    gemm(amsg, HD, NA, HD, woB_t, fexp, b_o, out_ah, nullptr, 1);
    seg_mean_k<<<NM, 256, 0, stream>>>(out_ah, atom2mol, NA, out_mol);
}
